// Block_1382979470189
// MI455X (gfx1250) — compile-verified, matrix operand order checked by reading, not run on hardware
//
#include <hip/hip_runtime.h>
#include <hip/hip_bf16.h>

// ---------------------------------------------------------------------------
// Types
// ---------------------------------------------------------------------------
typedef __bf16 bf16;
typedef bf16  v16bf __attribute__((ext_vector_type(16)));
typedef float v8f   __attribute__((ext_vector_type(8)));

#define TOKENS 16384
#define CDIM   384
#define QKVDIM 1152
#define HIDDEN 1536
#define NHEAD  6
#define HDIM   64
#define NSEQ   2048

// ---------------------------------------------------------------------------
// CDNA5 async global->LDS copies (ASYNCcnt-tracked).
// The clang builtin takes (int4 AS1* global, int4 AS3* lds, imm offset, imm cpol)
// per the round-2 diagnostic. Fall back to raw ISA asm if the builtin is absent.
// ---------------------------------------------------------------------------
#if __has_builtin(__builtin_amdgcn_global_load_async_to_lds_b128)
#define ASYNC_MODE 2
#else
#define ASYNC_MODE 1
#endif

typedef int v4i __attribute__((ext_vector_type(4)));
typedef __attribute__((address_space(1))) v4i* gv4i_p;
typedef __attribute__((address_space(3))) v4i* lv4i_p;

__device__ __forceinline__ void async_copy16(const bf16* g, bf16* l) {
#if ASYNC_MODE == 2
  __builtin_amdgcn_global_load_async_to_lds_b128(
      (gv4i_p)(uintptr_t)g, (lv4i_p)(unsigned int)(uintptr_t)l, 0, 0);
#else
  unsigned int laddr = (unsigned int)(uintptr_t)l;      // generic LDS addr[31:0] = LDS offset
  unsigned long long gaddr = (unsigned long long)(uintptr_t)g;
  asm volatile("global_load_async_to_lds_b128 %0, %1, off"
               :: "v"(laddr), "v"(gaddr) : "memory");
#endif
}

template <int N>
__device__ __forceinline__ void wait_async() {
#if (ASYNC_MODE == 2) && __has_builtin(__builtin_amdgcn_s_wait_asynccnt)
  __builtin_amdgcn_s_wait_asynccnt(N);
#else
  asm volatile("s_wait_asynccnt %0" :: "i"(N) : "memory");
#endif
}

// ---------------------------------------------------------------------------
// Fragment loaders (ISA 7.12.2 layouts, wave32).
// A fragment 16x32 bf16: lane l holds row M=l%16; element e holds
//   K = (e&8)*2 + ((l>=16)?8:0) + (e&7)   -> two contiguous 16B runs per lane,
// which the compiler folds into ds_load_b128 pairs.
// ---------------------------------------------------------------------------
__device__ __forceinline__ v16bf load_frag16(const bf16* __restrict__ tile,
                                             int row0, int stride, int k0, int lane) {
  const int m  = lane & 15;
  const int kh = (lane >> 4) << 3;
  const bf16* p = tile + (row0 + m) * stride + k0;
  v16bf f;
#pragma unroll
  for (int e = 0; e < 16; ++e) {
    int k = ((e & 8) << 1) + kh + (e & 7);
    f[e] = p[k];
  }
  return f;
}

// B fragment where the LDS tile is stored K-major: tile[k][n]
__device__ __forceinline__ v16bf load_frag16_kmaj(const bf16* __restrict__ tile,
                                                  int k0, int stride, int n0, int lane) {
  const int n  = lane & 15;
  const int kh = (lane >> 4) << 3;
  v16bf f;
#pragma unroll
  for (int e = 0; e < 16; ++e) {
    int k = ((e & 8) << 1) + kh + (e & 7);
    f[e] = tile[(k0 + k) * stride + n0 + n];
  }
  return f;
}

// ---------------------------------------------------------------------------
// fp32 -> bf16 cast (weights)
// ---------------------------------------------------------------------------
__global__ void cast_f32_to_bf16(const float* __restrict__ in, bf16* __restrict__ out, int n) {
  int i = blockIdx.x * blockDim.x + threadIdx.x;
  if (i < n) out[i] = (bf16)in[i];
}

// ---------------------------------------------------------------------------
// LayerNorm over 384 features; one wave per token; bf16 output
// ---------------------------------------------------------------------------
__global__ void ln_kernel(const float* __restrict__ x, const float* __restrict__ w,
                          const float* __restrict__ b, bf16* __restrict__ out) {
  const int lane  = threadIdx.x & 31;
  const int wave  = threadIdx.x >> 5;
  const int token = blockIdx.x * 4 + wave;
  const float* px = x + (size_t)token * CDIM;
  float v[12];
  float s = 0.f;
#pragma unroll
  for (int i = 0; i < 12; ++i) { v[i] = px[lane + i * 32]; s += v[i]; }
#pragma unroll
  for (int msk = 16; msk >= 1; msk >>= 1) s += __shfl_xor(s, msk, 32);
  const float mu = s * (1.0f / (float)CDIM);
  float q = 0.f;
#pragma unroll
  for (int i = 0; i < 12; ++i) { float d = v[i] - mu; q += d * d; }
#pragma unroll
  for (int msk = 16; msk >= 1; msk >>= 1) q += __shfl_xor(q, msk, 32);
  const float rstd = rsqrtf(q * (1.0f / (float)CDIM) + 1e-5f);
  bf16* po = out + (size_t)token * CDIM;
#pragma unroll
  for (int i = 0; i < 12; ++i) {
    int c = lane + i * 32;
    po[c] = (bf16)((v[i] - mu) * rstd * w[c] + b[c]);
  }
}

// ---------------------------------------------------------------------------
// bf16 WMMA GEMM:  C[M,N] = A[M,K] @ W[N,K]^T  (+bias) (+gelu) (+ls residual)
// Block 128 threads (4 waves). WG tile 128x64, K-step 32, double-buffered LDS
// filled with async global->LDS (next tile fetched while WMMAs run).
// Wave w computes rows [bm+32w, +32) x cols [bn, +64): 8 WMMAs / K-step.
// ---------------------------------------------------------------------------
__global__ void __launch_bounds__(128)
gemm_wmma(const bf16* __restrict__ A, const bf16* __restrict__ W,
          const float* __restrict__ bias, const float* __restrict__ gamma,
          const float* __restrict__ resid, float* __restrict__ outF,
          bf16* __restrict__ outB, int N, int K, int act) {
  __shared__ bf16 sA[2][128][48];
  __shared__ bf16 sW[2][64][48];
  const int tid  = threadIdx.x;
  const int lane = tid & 31;
  const int wave = tid >> 5;
  const int bn = blockIdx.x * 64;
  const int bm = blockIdx.y * 128;

  const v8f vzero = {0.f, 0.f, 0.f, 0.f, 0.f, 0.f, 0.f, 0.f};
  v8f acc[2][4];
#pragma unroll
  for (int mt = 0; mt < 2; ++mt)
#pragma unroll
    for (int nt = 0; nt < 4; ++nt) acc[mt][nt] = vzero;

  // issue one K-tile (A: 128x32, W: 64x32) into buffer b — 6 asyncs / thread
  auto issue = [&](int k0, int b) {
#pragma unroll
    for (int j = 0; j < 4; ++j) {
      int c = j * 128 + tid;               // 0..511 chunks of 8 bf16
      int row = c >> 2, col = (c & 3) * 8;
      async_copy16(A + (size_t)(bm + row) * K + k0 + col, &sA[b][row][col]);
    }
#pragma unroll
    for (int j = 0; j < 2; ++j) {
      int c = j * 128 + tid;               // 0..255
      int row = c >> 2, col = (c & 3) * 8;
      async_copy16(W + (size_t)(bn + row) * K + k0 + col, &sW[b][row][col]);
    }
  };

  issue(0, 0);
  int ib = 0;
  for (int k0 = 0; k0 < K; k0 += 32, ib ^= 1) {
    if (k0 + 32 < K) { issue(k0 + 32, ib ^ 1); wait_async<6>(); }
    else             { wait_async<0>(); }
    __syncthreads();

    v16bf a0 = load_frag16(&sA[ib][0][0], wave * 32,      48, 0, lane);
    v16bf a1 = load_frag16(&sA[ib][0][0], wave * 32 + 16, 48, 0, lane);
#pragma unroll
    for (int nt = 0; nt < 4; ++nt) {
      v16bf bfrag = load_frag16(&sW[ib][0][0], nt * 16, 48, 0, lane);
      acc[0][nt] = __builtin_amdgcn_wmma_f32_16x16x32_bf16(
          false, a0, false, bfrag, (short)0, acc[0][nt], false, false);
      acc[1][nt] = __builtin_amdgcn_wmma_f32_16x16x32_bf16(
          false, a1, false, bfrag, (short)0, acc[1][nt], false, false);
    }
    __syncthreads();
  }

  // epilogue: lane l holds col n=l%16 (per n-tile), rows e + 8*(l>=16)
  const int col0 = bn + (lane & 15);
#pragma unroll
  for (int mt = 0; mt < 2; ++mt) {
    const int rowb = bm + wave * 32 + mt * 16 + ((lane >> 4) << 3);
#pragma unroll
    for (int nt = 0; nt < 4; ++nt) {
      const int col = col0 + nt * 16;
      const float bia = bias ? bias[col] : 0.f;
#pragma unroll
      for (int e = 0; e < 8; ++e) {
        const int row = rowb + e;
        float v = acc[mt][nt][e] + bia;
        if (act == 1) v = 0.5f * v * (1.0f + erff(v * 0.70710678118f));
        if (outF) {
          outF[(size_t)row * N + col] = resid[(size_t)row * N + col] + gamma[col] * v;
        } else {
          outB[(size_t)row * N + col] = (bf16)v;
        }
      }
    }
  }
}

// ---------------------------------------------------------------------------
// Flash attention, bf16 WMMA, fp32 accum & online softmax.
// Grid: (NSEQ/64, NHEAD, B); block 128 (4 waves); wave owns a 16-row Q strip.
// Key tile = 32 so P(16x32) is exactly one bf16 A-fragment.
// K/V tiles double-buffered via async global->LDS.
// ---------------------------------------------------------------------------
__global__ void __launch_bounds__(128)
attn_wmma(const bf16* __restrict__ qkv, bf16* __restrict__ out) {
  __shared__ bf16 sQ[64][72];
  __shared__ bf16 sK[2][32][72];
  __shared__ bf16 sV[2][32][72];
  __shared__ bf16 sP[4][16][40];

  const int tid = threadIdx.x, lane = tid & 31, wave = tid >> 5;
  const int h     = blockIdx.y;
  const int bbase = blockIdx.z * NSEQ;
  const int m0    = blockIdx.x * 64;

  const int kvrow = tid >> 2;        // 0..31
  const int kvch  = (tid & 3) * 16;  // 0,16,32,48

  // issue one 32-key K/V tile into buffer b — 4 asyncs / thread
  auto issue_kv = [&](int kt, int b) {
    const bf16* srck = qkv + (size_t)(bbase + kt + kvrow) * QKVDIM + CDIM     + h * HDIM + kvch;
    const bf16* srcv = qkv + (size_t)(bbase + kt + kvrow) * QKVDIM + 2 * CDIM + h * HDIM + kvch;
    async_copy16(srck,     &sK[b][kvrow][kvch]);
    async_copy16(srck + 8, &sK[b][kvrow][kvch + 8]);
    async_copy16(srcv,     &sV[b][kvrow][kvch]);
    async_copy16(srcv + 8, &sV[b][kvrow][kvch + 8]);
  };

  issue_kv(0, 0);

  // ---- load Q tile 64x64 (once, synchronous) ----
  {
    const int row = tid >> 1;
    const int ch  = (tid & 1) * 32;
    const bf16* src = qkv + (size_t)(bbase + m0 + row) * QKVDIM + h * HDIM + ch;
#pragma unroll
    for (int j = 0; j < 32; j += 8)
      *(uint4*)(&sQ[row][ch + j]) = *(const uint4*)(src + j);
  }
  __syncthreads();
  v16bf qf[2];
  qf[0] = load_frag16(&sQ[0][0], wave * 16, 72, 0, lane);
  qf[1] = load_frag16(&sQ[0][0], wave * 16, 72, 32, lane);

  const v8f vzero = {0.f, 0.f, 0.f, 0.f, 0.f, 0.f, 0.f, 0.f};
  v8f o[4];
  float mrow[8], lrow[8];
#pragma unroll
  for (int nt = 0; nt < 4; ++nt) o[nt] = vzero;
#pragma unroll
  for (int r = 0; r < 8; ++r) { mrow[r] = -1e30f; lrow[r] = 0.f; }
  const float scale = 0.125f;  // HDIM^-0.5

  int ib = 0;
  for (int kt = 0; kt < NSEQ; kt += 32, ib ^= 1) {
    if (kt + 32 < NSEQ) { issue_kv(kt + 32, ib ^ 1); wait_async<4>(); }
    else                { wait_async<0>(); }
    __syncthreads();

    // S = Q @ K^T : 16x32 scores as two 16x16 accumulators
    v8f s[2];
    s[0] = vzero; s[1] = vzero;
#pragma unroll
    for (int nt = 0; nt < 2; ++nt) {
      v16bf b0 = load_frag16(&sK[ib][0][0], nt * 16, 72, 0, lane);
      s[nt] = __builtin_amdgcn_wmma_f32_16x16x32_bf16(
          false, qf[0], false, b0, (short)0, s[nt], false, false);
      v16bf b1 = load_frag16(&sK[ib][0][0], nt * 16, 72, 32, lane);
      s[nt] = __builtin_amdgcn_wmma_f32_16x16x32_bf16(
          false, qf[1], false, b1, (short)0, s[nt], false, false);
    }

    // online softmax (stats per row live in 16-lane halves)
#pragma unroll
    for (int r = 0; r < 8; ++r) {
      float v0 = s[0][r] * scale, v1 = s[1][r] * scale;
      float mx = fmaxf(v0, v1);
#pragma unroll
      for (int msk = 8; msk >= 1; msk >>= 1) mx = fmaxf(mx, __shfl_xor(mx, msk, 32));
      const float mnew  = fmaxf(mrow[r], mx);
      const float alpha = __expf(mrow[r] - mnew);
      mrow[r] = mnew;
      const float p0 = __expf(v0 - mnew);
      const float p1 = __expf(v1 - mnew);
      float ps = p0 + p1;
#pragma unroll
      for (int msk = 8; msk >= 1; msk >>= 1) ps += __shfl_xor(ps, msk, 32);
      lrow[r] = lrow[r] * alpha + ps;
#pragma unroll
      for (int nt = 0; nt < 4; ++nt) o[nt][r] *= alpha;
      const int rr = r + ((lane >> 4) << 3);
      sP[wave][rr][lane & 15]        = (bf16)p0;
      sP[wave][rr][16 + (lane & 15)] = (bf16)p1;
    }

    // O += P @ V  (P: 16x32 A-frag, V: 32x64 K-major)
    v16bf pf = load_frag16(&sP[wave][0][0], 0, 40, 0, lane);
#pragma unroll
    for (int nt = 0; nt < 4; ++nt) {
      v16bf vf = load_frag16_kmaj(&sV[ib][0][0], 0, 72, nt * 16, lane);
      o[nt] = __builtin_amdgcn_wmma_f32_16x16x32_bf16(
          false, pf, false, vf, (short)0, o[nt], false, false);
    }
    __syncthreads();
  }

  // write O / l  ->  [B,N,C] bf16 (head h occupies cols h*64..h*64+63)
#pragma unroll
  for (int nt = 0; nt < 4; ++nt) {
#pragma unroll
    for (int e = 0; e < 8; ++e) {
      const int row = m0 + wave * 16 + e + ((lane >> 4) << 3);
      const float v = o[nt][e] / lrow[e];
      out[(size_t)(bbase + row) * CDIM + h * HDIM + nt * 16 + (lane & 15)] = (bf16)v;
    }
  }
}

// ---------------------------------------------------------------------------
// Host-side orchestration
// ---------------------------------------------------------------------------
extern "C" void kernel_launch(void* const* d_in, const int* in_sizes, int n_in,
                              void* d_out, int out_size, void* d_ws, size_t ws_size,
                              hipStream_t stream) {
  (void)in_sizes; (void)n_in; (void)out_size; (void)ws_size;
  const float* x       = (const float*)d_in[0];
  const float* qkv_w   = (const float*)d_in[1];
  const float* qkv_b   = (const float*)d_in[2];
  const float* proj_w  = (const float*)d_in[3];
  const float* proj_b  = (const float*)d_in[4];
  const float* fc1_w   = (const float*)d_in[5];
  const float* fc1_b   = (const float*)d_in[6];
  const float* eye1_w  = (const float*)d_in[7];
  const float* eye2_w  = (const float*)d_in[8];
  const float* fc2_w   = (const float*)d_in[9];
  const float* fc2_b   = (const float*)d_in[10];
  const float* norm1_w = (const float*)d_in[11];
  const float* norm1_b = (const float*)d_in[12];
  const float* norm2_w = (const float*)d_in[13];
  const float* norm2_b = (const float*)d_in[14];
  const float* ls1     = (const float*)d_in[15];
  const float* ls2     = (const float*)d_in[16];

  char* ws = (char*)d_ws;
  size_t off = 0;
  auto alloc = [&](size_t bytes) -> void* {
    void* p = ws + off;
    off += (bytes + 255) & ~(size_t)255;
    return p;
  };

  const size_t T = TOKENS;
  bf16* wq  = (bf16*)alloc((size_t)QKVDIM * CDIM * 2);
  bf16* wp  = (bf16*)alloc((size_t)CDIM * CDIM * 2);
  bf16* w1  = (bf16*)alloc((size_t)HIDDEN * CDIM * 2);
  bf16* we1 = (bf16*)alloc((size_t)HIDDEN * HIDDEN * 2);
  bf16* we2 = (bf16*)alloc((size_t)HIDDEN * HIDDEN * 2);
  bf16* w2  = (bf16*)alloc((size_t)CDIM * HIDDEN * 2);
  bf16* hb   = (bf16*)alloc(T * CDIM * 2);    // LN1 out  (reused later as a3)
  bf16* qkvb = (bf16*)alloc(T * QKVDIM * 2);  // QKV out  (contiguous after hb)
  bf16* ao   = (bf16*)alloc(T * CDIM * 2);    // attention out
  float* x1  = (float*)alloc(T * CDIM * 4);   // x + ls1*proj
  bf16* h2   = (bf16*)alloc(T * CDIM * 2);    // LN2 out
  bf16* a1   = (bf16*)alloc(T * HIDDEN * 2);  // fc1 out
  bf16* a2   = (bf16*)alloc(T * HIDDEN * 2);  // eye1 out
  bf16* a3   = hb;                            // eye2 out aliases hb+qkvb (exact fit)

  auto cast = [&](const float* src, bf16* dst, int n) {
    cast_f32_to_bf16<<<(n + 255) / 256, 256, 0, stream>>>(src, dst, n);
  };
  cast(qkv_w, wq, QKVDIM * CDIM);
  cast(proj_w, wp, CDIM * CDIM);
  cast(fc1_w, w1, HIDDEN * CDIM);
  cast(eye1_w, we1, HIDDEN * HIDDEN);
  cast(eye2_w, we2, HIDDEN * HIDDEN);
  cast(fc2_w, w2, CDIM * HIDDEN);

  const dim3 blk(128);

  // LN1
  ln_kernel<<<T / 4, 128, 0, stream>>>(x, norm1_w, norm1_b, hb);
  // QKV GEMM  [T,1152] = h @ qkv_w^T + b
  gemm_wmma<<<dim3(QKVDIM / 64, T / 128), blk, 0, stream>>>(
      hb, wq, qkv_b, nullptr, nullptr, nullptr, qkvb, QKVDIM, CDIM, 0);
  // attention
  attn_wmma<<<dim3(NSEQ / 64, NHEAD, 8), blk, 0, stream>>>(qkvb, ao);
  // proj + layer-scale residual -> x1 (fp32)
  gemm_wmma<<<dim3(CDIM / 64, T / 128), blk, 0, stream>>>(
      ao, wp, proj_b, ls1, x, x1, nullptr, CDIM, CDIM, 0);
  // LN2
  ln_kernel<<<T / 4, 128, 0, stream>>>(x1, norm2_w, norm2_b, h2);
  // fc1 + GELU
  gemm_wmma<<<dim3(HIDDEN / 64, T / 128), blk, 0, stream>>>(
      h2, w1, fc1_b, nullptr, nullptr, nullptr, a1, HIDDEN, CDIM, 1);
  // eye1
  gemm_wmma<<<dim3(HIDDEN / 64, T / 128), blk, 0, stream>>>(
      a1, we1, nullptr, nullptr, nullptr, nullptr, a2, HIDDEN, HIDDEN, 0);
  // eye2
  gemm_wmma<<<dim3(HIDDEN / 64, T / 128), blk, 0, stream>>>(
      a2, we2, nullptr, nullptr, nullptr, nullptr, a3, HIDDEN, HIDDEN, 0);
  // fc2 + layer-scale residual -> d_out (fp32)
  gemm_wmma<<<dim3(CDIM / 64, T / 128), blk, 0, stream>>>(
      a3, w2, fc2_b, ls2, x1, (float*)d_out, nullptr, CDIM, HIDDEN, 0);
}
